// MatchModel_63531156242905
// MI455X (gfx1250) — 16-bit and 8-bit matrix operand order repaired and checked by reading, compiled, not run on hardware
//
#include <hip/hip_runtime.h>
#include <hip/hip_bf16.h>

typedef __attribute__((ext_vector_type(2)))  float    v2f;
typedef __attribute__((ext_vector_type(8)))  float    v8f;
typedef __attribute__((ext_vector_type(8)))  int      v8i;
typedef __attribute__((ext_vector_type(16))) _Float16 v16h;

namespace {
constexpr int   PN = 256, DN = 256, TN = 2, ON = 24, OPAD = 32;
constexpr int   HN = 240, WN = 432, HWN = HN * WN;
constexpr int   KB64 = HWN / 64;
constexpr float EPSF = 1e-8f, SW = 0.5f, LRF = 0.1f;
constexpr int   MAXIT = 20, PROJIT = 5;

constexpr size_t OFF_SUMA  = 0;
constexpr size_t OFF_SUMB  = 128;
constexpr size_t OFF_INTER = 1152;
constexpr size_t OFF_FS    = OFF_INTER + 32768;
constexpr size_t OFF_BINX  = OFF_FS + 32768;
constexpr size_t OFF_KF    = OFF_BINX + 32768;
constexpr size_t OFF_QFP   = OFF_KF + 262144;
constexpr size_t ZERO_BYTES = 458752;
constexpr size_t OFF_A8    = 458752;
constexpr size_t OFF_B8    = OFF_A8 + (size_t)OPAD * HWN;
constexpr size_t OFF_BT    = OFF_B8 + (size_t)PN * HWN;
constexpr size_t OUTM      = (size_t)ON * HWN;
}

__device__ __forceinline__ float wave_sum(float v) {
  #pragma unroll
  for (int off = 16; off; off >>= 1) v += __shfl_xor(v, off, 32);
  return v;
}
__device__ __forceinline__ float wave_max(float v) {
  #pragma unroll
  for (int off = 16; off; off >>= 1) v = fmaxf(v, __shfl_xor(v, off, 32));
  return v;
}

__global__ void zero_ws(unsigned* __restrict__ w) {
  w[(size_t)blockIdx.x * 256 + threadIdx.x] = 0u;
}

__global__ void pack_templates(const float* __restrict__ maskT,
                               unsigned char* __restrict__ A8,
                               float* __restrict__ sumA) {
  __shared__ float red[256];
  size_t idx = (size_t)blockIdx.x * 256 + threadIdx.x;
  int o = (int)(idx / HWN);
  int k = (int)(idx % HWN);
  unsigned char b = 0;
  if (o < ON) b = (maskT[(size_t)o * HWN + k] > 0.5f) ? 1 : 0;
  A8[idx] = b;
  red[threadIdx.x] = (float)b;
  __syncthreads();
  #pragma unroll
  for (int s = 128; s; s >>= 1) {
    if (threadIdx.x < s) red[threadIdx.x] += red[threadIdx.x + s];
    __syncthreads();
  }
  if (threadIdx.x == 0) atomicAdd(&sumA[o], red[0]);
}

__global__ void pack_proposals(const float* __restrict__ maskP,
                               unsigned char* __restrict__ B8,
                               unsigned char* __restrict__ Bt,
                               float* __restrict__ sumB) {
  __shared__ unsigned char lds[16][256];
  const int kbase = blockIdx.x * 256;
  const int pt    = blockIdx.y;
  const int r     = threadIdx.x >> 4;
  const int c0    = threadIdx.x & 15;
  const int p     = pt * 16 + r;
  float s = 0.f;
  #pragma unroll
  for (int j = 0; j < 16; ++j) {
    int k = kbase + c0 + j * 16;
    unsigned char b = (maskP[(size_t)p * HWN + k] > 0.5f) ? 1 : 0;
    lds[r][c0 + j * 16] = b;
    B8[(size_t)p * HWN + k] = b;
    s += (float)b;
  }
  atomicAdd(&sumB[p], s);
  __syncthreads();
  const int k = kbase + threadIdx.x;
  unsigned w[4];
  #pragma unroll
  for (int q = 0; q < 4; ++q) {
    w[q] = (unsigned)lds[4*q + 0][threadIdx.x]
         | ((unsigned)lds[4*q + 1][threadIdx.x] << 8)
         | ((unsigned)lds[4*q + 2][threadIdx.x] << 16)
         | ((unsigned)lds[4*q + 3][threadIdx.x] << 24);
  }
  uint4 v; v.x = w[0]; v.y = w[1]; v.z = w[2]; v.w = w[3];
  *(uint4*)(Bt + ((size_t)pt * HWN + (size_t)k) * 16) = v;
}

__global__ void normalize_features(const float* __restrict__ pf,
                                   const float* __restrict__ tf,
                                   float* __restrict__ kf,
                                   float* __restrict__ qfp) {
  int wave = blockIdx.x * 8 + (threadIdx.x >> 5);
  int lane = threadIdx.x & 31;
  const float* in; float* out;
  if (wave < PN) {
    in = pf + (size_t)wave * DN; out = kf + (size_t)wave * DN;
  } else if (wave < PN + TN * ON) {
    int i = wave - PN; int t = i / ON, o = i % ON;
    in = tf + (size_t)i * DN;
    out = qfp + (size_t)o * (TN * DN) + (size_t)t * DN;
  } else return;
  float v[8]; float ss = 0.f;
  #pragma unroll
  for (int j = 0; j < 8; ++j) { v[j] = in[lane + j * 32]; ss += v[j] * v[j]; }
  ss = wave_sum(ss);
  float sc = 1.0f / (sqrtf(ss) + EPSF);
  #pragma unroll
  for (int j = 0; j < 8; ++j) out[lane + j * 32] = v[j] * sc;
}

__global__ void __launch_bounds__(256) iou_inter_wmma(
    const unsigned char* __restrict__ A8,
    const unsigned char* __restrict__ Bt,
    int* __restrict__ inter) {
  const int wave  = blockIdx.x * 8 + (threadIdx.x >> 5);
  const int tile  = wave >> 3;
  const int slice = wave & 7;
  const int ot = tile >> 4, pt = tile & 15;
  const int lane = threadIdx.x & 31;
  const int m = lane & 15, h = lane >> 4;

  const unsigned char* Arow = A8 + (size_t)(ot * 16 + m) * HWN + h * 8;
  const unsigned char* Bcol = Bt + (size_t)pt * HWN * 16 + m;

  v8i acc = {};
  for (int kb = slice; kb < KB64; kb += 8) {
    const unsigned char* ap = Arow + kb * 64;
    int2 a01 = *(const int2*)(ap +  0);
    int2 a23 = *(const int2*)(ap + 16);
    int2 a45 = *(const int2*)(ap + 32);
    int2 a67 = *(const int2*)(ap + 48);
    v8i a; a[0]=a01.x; a[1]=a01.y; a[2]=a23.x; a[3]=a23.y;
           a[4]=a45.x; a[5]=a45.y; a[6]=a67.x; a[7]=a67.y;
    const unsigned char* bp = Bcol + (size_t)kb * (64 * 16);
    v8i b = {};
    #pragma unroll
    for (int s = 0; s < 32; ++s)
      b[s >> 2] |= (int)bp[(size_t)(((s >> 3) << 4) + h * 8 + (s & 7)) * 16] << (8 * (s & 3));
    __builtin_prefetch(bp + (size_t)8 * 64 * 16, 0, 1);
    acc = __builtin_amdgcn_wmma_i32_16x16x64_iu8(false, a, false, b, acc, false, false);
  }
  const int col = pt * 16 + m;
  #pragma unroll
  for (int j = 0; j < 8; ++j) {
    int row = ot * 16 + j + (h ? 8 : 0);
    atomicAdd(&inter[row * 256 + col], acc[j]);
  }
}

__global__ void __launch_bounds__(256) feat_sim_wmma(
    const float* __restrict__ kf, const float* __restrict__ qfp,
    float* __restrict__ fs) {
  const int wave = blockIdx.x * 8 + (threadIdx.x >> 5);
  const int ot = wave >> 4, pt = wave & 15;
  const int lane = threadIdx.x & 31;
  const int m = lane & 15, h = lane >> 4;
  const float* arow = qfp + (size_t)(ot * 16 + m) * 512;
  const float* brow = kf  + (size_t)(pt * 16 + m) * 256;
  v8f acc = {};
#if __has_builtin(__builtin_amdgcn_wmma_f32_16x16x4_f32)
  for (int k = 0; k < 512; k += 4) {
    v2f a, b;
    a[0] = arow[k + 2 * h]; a[1] = arow[k + 2 * h + 1];
    int d = (k & 255) + 2 * h;
    b[0] = brow[d]; b[1] = brow[d + 1];
    acc = __builtin_amdgcn_wmma_f32_16x16x4_f32(false, a, false, b,
                                                (short)0, acc, false, false);
  }
#else
  #pragma unroll 1
  for (int j = 0; j < 8; ++j) {
    const float* ar = qfp + (size_t)(ot * 16 + j + (h ? 8 : 0)) * 512;
    float s = 0.f;
    for (int k = 0; k < 512; ++k) s += ar[k] * brow[k & 255];
    acc[j] = s;
  }
#endif
  const int col = pt * 16 + m;
  #pragma unroll
  for (int j = 0; j < 8; ++j) {
    int row = ot * 16 + j + (h ? 8 : 0);
    fs[row * 256 + col] = acc[j] * (1.0f / (float)TN);
  }
}

__global__ void __launch_bounds__(768) relax_match(
    const int* __restrict__ inter, const float* __restrict__ fs,
    const float* __restrict__ sumA, const float* __restrict__ sumB,
    const float* __restrict__ score, float* __restrict__ binX,
    float* __restrict__ outScores) {
  const int o = threadIdx.x >> 5;
  const int lane = threadIdx.x & 31;
  float C[8], S[8], X[8], XS[8];
  const float sa = sumA[o];
  #pragma unroll
  for (int j = 0; j < 8; ++j) {
    int n = lane + j * 32;
    float it = (float)inter[o * 256 + n];
    float u  = sa + sumB[n] - it;
    float iou = it / (u + EPSF);
    float s = (1.0f - SW) * fs[o * 256 + n] + SW * iou;
    S[j] = s; C[j] = -s;
    X[j] = 1.0f / 256.0f; XS[j] = 0.f;
  }
  for (int it = 0; it < MAXIT; ++it) {
    #pragma unroll
    for (int j = 0; j < 8; ++j) X[j] -= LRF * C[j];
    for (int pi = 0; pi < PROJIT; ++pi) {
      float local = 0.f;
      #pragma unroll
      for (int j = 0; j < 8; ++j) {
        X[j] = fminf(fmaxf(X[j], 0.f), 1.f);
        local += X[j];
      }
      float inv = 1.0f / (wave_sum(local) + EPSF);
      #pragma unroll
      for (int j = 0; j < 8; ++j) X[j] *= inv;
    }
    #pragma unroll
    for (int j = 0; j < 8; ++j) XS[j] += X[j];
  }
  float ms = -1e30f, ds = 0.f;
  #pragma unroll
  for (int j = 0; j < 8; ++j) {
    int n = lane + j * 32;
    float xm = XS[j] * (1.0f / (float)MAXIT);
    float bx = (xm > 0.01f) ? xm : 0.f;
    binX[o * 256 + n] = bx;
    ms = fmaxf(ms, fminf(fmaxf(xm, 0.f), 1.f) * S[j]);
    ds += score[n] * bx;
  }
  ms = wave_max(ms);
  ds = wave_sum(ds);
  if (lane == 0) { outScores[o] = ms; outScores[ON + o] = ds; }
}

__global__ void __launch_bounds__(256) outmask_wmma(
    const float* __restrict__ binX, const unsigned char* __restrict__ B8,
    float* __restrict__ outMask) {
  const int wave = blockIdx.x * 8 + (threadIdx.x >> 5);
  const int ot  = wave / (HWN / 16);
  const int nt  = wave % (HWN / 16);
  const int hw0 = nt * 16;
  const int lane = threadIdx.x & 31;
  const int m = lane & 15, h = lane >> 4;
  const float* arow = binX + (size_t)(ot * 16 + m) * 256;
  v8f acc = {};
  #pragma unroll 1
  for (int kb = 0; kb < 8; ++kb) {
    v16h a, b;
    const float* ap = arow + kb * 32 + h * 8;
    #pragma unroll
    for (int i = 0; i < 8; ++i) {
      a[i]     = (_Float16)ap[i];
      a[8 + i] = (_Float16)ap[16 + i];
    }
    const unsigned char* bp = B8 + (size_t)(kb * 32 + h * 8) * HWN + hw0 + m;
    #pragma unroll
    for (int i = 0; i < 8; ++i) {
      b[i]     = (_Float16)(float)bp[(size_t)i * HWN];
      b[8 + i] = (_Float16)(float)bp[(size_t)(16 + i) * HWN];
    }
    __builtin_prefetch(bp + (size_t)32 * HWN, 0, 1);
    acc = __builtin_amdgcn_wmma_f32_16x16x32_f16(false, a, false, b,
                                                 (short)0, acc, false, false);
  }
  #pragma unroll
  for (int j = 0; j < 8; ++j) {
    int row = ot * 16 + j + (h ? 8 : 0);
    if (row < ON) outMask[(size_t)row * HWN + hw0 + m] = acc[j];
  }
}

extern "C" void kernel_launch(void* const* d_in, const int* in_sizes, int n_in,
                              void* d_out, int out_size, void* d_ws, size_t ws_size,
                              hipStream_t stream) {
  const float* pf = (const float*)d_in[0];
  const float* pm = (const float*)d_in[1];
  const float* tf = (const float*)d_in[2];
  const float* ml = (const float*)d_in[3];
  const float* sc = (const float*)d_in[4];
  float* out = (float*)d_out;
  char* ws = (char*)d_ws;
  float* sumA = (float*)(ws + OFF_SUMA);
  float* sumB = (float*)(ws + OFF_SUMB);
  int*   inter= (int*)(ws + OFF_INTER);
  float* fs   = (float*)(ws + OFF_FS);
  float* binX = (float*)(ws + OFF_BINX);
  float* kf   = (float*)(ws + OFF_KF);
  float* qfp  = (float*)(ws + OFF_QFP);
  unsigned char* A8 = (unsigned char*)(ws + OFF_A8);
  unsigned char* B8 = (unsigned char*)(ws + OFF_B8);
  unsigned char* Bt = (unsigned char*)(ws + OFF_BT);

  zero_ws<<<(int)(ZERO_BYTES / 1024), 256, 0, stream>>>((unsigned*)ws);
  pack_templates<<<(OPAD * HWN) / 256, 256, 0, stream>>>(ml, A8, sumA);
  pack_proposals<<<dim3(HWN / 256, 16), 256, 0, stream>>>(pm, B8, Bt, sumB);
  normalize_features<<<(PN + TN * ON + 7) / 8, 256, 0, stream>>>(pf, tf, kf, qfp);
  iou_inter_wmma<<<32, 256, 0, stream>>>(A8, Bt, inter);
  feat_sim_wmma<<<4, 256, 0, stream>>>(kf, qfp, fs);
  relax_match<<<1, 768, 0, stream>>>(inter, fs, sumA, sumB, sc, binX, out + OUTM);
  outmask_wmma<<<(2 * (HWN / 16)) / 8, 256, 0, stream>>>(binX, B8, out);
}
